// SparseConvNeXtV2_34935263986142
// MI455X (gfx1250) — compile-verified, matrix operand order checked by reading, not run on hardware
//
#include <hip/hip_runtime.h>
#include <hip/hip_bf16.h>
#include <math.h>

// ---------------------------------------------------------------------------
// Sparse ConvNeXtV2 forward (FCMAE-style masking) for gfx1250 / MI455X.
// Pointwise GEMMs use v_wmma_f32_16x16x32_bf16 (bf16 in, f32 accumulate).
// LDS tiles are stored pre-packed as bf16x2 dwords so each WMMA fragment is
// two ds_load_b128 ops; each wave owns a 32x32 output tile (4 WMMA/K-step).
// The K-loop is cloned on the wave-uniform "active" predicate so accumulators
// stay in-place across iterations (no cross-branch phis / register copies).
// Elementwise / dwconv / LN / GRN stages are bandwidth-bound wave32 VALU code.
// ---------------------------------------------------------------------------

typedef __bf16    v16bf __attribute__((ext_vector_type(16)));
typedef float     v8f   __attribute__((ext_vector_type(8)));
typedef unsigned  v8u   __attribute__((ext_vector_type(8)));

__device__ __forceinline__ float wred(float v) {
  #pragma unroll
  for (int o = 16; o > 0; o >>= 1) v += __shfl_xor(v, o, 32);
  return v;
}

__device__ __forceinline__ float gelu_exact(float v) {
  return 0.5f * v * (1.0f + erff(v * 0.70710678118654752f));
}

__device__ __forceinline__ unsigned pack_bf2(float lo, float hi) {
  union { __bf16 h[2]; unsigned u; } p;
  p.h[0] = (__bf16)lo;   // -> v_cvt_pk_bf16_f32
  p.h[1] = (__bf16)hi;
  return p.u;
}

// Build a 16x32-element bf16 WMMA fragment from a packed-pair LDS row.
// Fragment dword j (j<4) holds k = kb+2j,kb+2j+1 -> pair index kb2+j;
// j>=4 holds k = 16+kb+2(j-4)... -> pair index 8+kb2+(j-4).  Two b128 loads.
__device__ __forceinline__ v16bf frag(const unsigned* rowp, int kb2) {
  uint4 lo = *(const uint4*)(rowp + kb2);
  uint4 hi = *(const uint4*)(rowp + 8 + kb2);
  v8u u = {lo.x, lo.y, lo.z, lo.w, hi.x, hi.y, hi.z, hi.w};
  return __builtin_bit_cast(v16bf, u);
}

// ----------------------------- mask kernels --------------------------------

__global__ __launch_bounds__(256) void build_mask_k(const int* __restrict__ mask,
                                                    float* __restrict__ m56) {
  int idx = blockIdx.x * 256 + threadIdx.x;
  if (idx >= 32 * 56 * 56) return;
  int x = idx % 56, y = (idx / 56) % 56, b = idx / (56 * 56);
  m56[idx] = 1.0f - (float)mask[b * 49 + (y >> 3) * 7 + (x >> 3)];
}

__global__ __launch_bounds__(256) void mask_down_k(const float* __restrict__ mi,
                                                   float* __restrict__ mo, int Sout) {
  int total = 32 * Sout * Sout;
  int idx = blockIdx.x * 256 + threadIdx.x;
  if (idx >= total) return;
  int x = idx % Sout, y = (idx / Sout) % Sout, b = idx / (Sout * Sout);
  int Sin = Sout * 2;
  const float* p = &mi[((size_t)b * Sin + 2 * y) * Sin + 2 * x];
  mo[idx] = fmaxf(fmaxf(p[0], p[1]), fmaxf(p[Sin], p[Sin + 1]));
}

// ------------------------------- stem --------------------------------------
// 4x4 s4 conv 3->96 (+bias), LayerNorm over C, mask. One wave per site,
// each lane owns 3 channels; LN stats via wave32 shuffle reductions.

__global__ __launch_bounds__(256) void stem_kernel(
    const float* __restrict__ in, const float* __restrict__ w,
    const float* __restrict__ bias, const float* __restrict__ g,
    const float* __restrict__ b, const float* __restrict__ m,
    float* __restrict__ out) {
  int wv = threadIdx.x >> 5, lane = threadIdx.x & 31;
  int site = blockIdx.x * 8 + wv;
  if (site >= 32 * 56 * 56) return;
  int px = site % 56, py = (site / 56) % 56, bb = site / (56 * 56);
  float vals[3];
  #pragma unroll
  for (int t = 0; t < 3; ++t) {
    int o = lane + 32 * t;
    float acc = bias[o];
    for (int ky = 0; ky < 4; ++ky)
      for (int kx = 0; kx < 4; ++kx)
        for (int i = 0; i < 3; ++i)
          acc += in[(((size_t)bb * 3 + i) * 224 + (4 * py + ky)) * 224 + (4 * px + kx)]
               * w[((ky * 4 + kx) * 3 + i) * 96 + o];
    vals[t] = acc;
  }
  float s  = vals[0] + vals[1] + vals[2];
  float s2 = vals[0]*vals[0] + vals[1]*vals[1] + vals[2]*vals[2];
  s = wred(s); s2 = wred(s2);
  float mean = s * (1.0f / 96.0f);
  float var  = s2 * (1.0f / 96.0f) - mean * mean;
  float inv  = rsqrtf(var + 1e-6f);
  float mv   = m[site];
  #pragma unroll
  for (int t = 0; t < 3; ++t) {
    int o = lane + 32 * t;
    out[(size_t)site * 96 + o] = ((vals[t] - mean) * inv * g[o] + b[o]) * mv;
  }
}

// --------------------------- depthwise 7x7 ---------------------------------

__global__ __launch_bounds__(256) void dwconv_mask(
    const float* __restrict__ in, float* __restrict__ out,
    const float* __restrict__ k, const float* __restrict__ kb,
    const float* __restrict__ m, int S, int C) {
  size_t idx = (size_t)blockIdx.x * 256 + threadIdx.x;
  size_t total = (size_t)32 * S * S * C;
  if (idx >= total) return;
  int c = (int)(idx % C);
  int site = (int)(idx / C);
  int x = site % S, y = (site / S) % S, b = site / (S * S);
  float acc = 0.0f;
  for (int ky = 0; ky < 7; ++ky) {
    int yy = y + ky - 3;
    if (yy < 0 || yy >= S) continue;
    for (int kx = 0; kx < 7; ++kx) {
      int xx = x + kx - 3;
      if (xx < 0 || xx >= S) continue;
      acc += in[(((size_t)b * S + yy) * S + xx) * C + c] * k[(ky * 7 + kx) * C + c];
    }
  }
  out[idx] = (acc + kb[c]) * m[site];
}

// ------------------------------ LayerNorm ----------------------------------

__global__ __launch_bounds__(256) void ln_mask(
    const float* __restrict__ in, float* __restrict__ out,
    const float* __restrict__ g, const float* __restrict__ b,
    const float* __restrict__ m, int N, int C) {
  int wv = threadIdx.x >> 5, lane = threadIdx.x & 31;
  int site = blockIdx.x * 8 + wv;
  if (site >= N) return;
  float s = 0.0f, s2 = 0.0f;
  for (int c = lane; c < C; c += 32) {
    float v = in[(size_t)site * C + c];
    s += v; s2 += v * v;
  }
  s = wred(s); s2 = wred(s2);
  float mean = s / (float)C;
  float var  = s2 / (float)C - mean * mean;
  float inv  = rsqrtf(var + 1e-6f);
  float mv   = m[site];
  for (int c = lane; c < C; c += 32) {
    float v = in[(size_t)site * C + c];
    out[(size_t)site * C + c] = ((v - mean) * inv * g[c] + b[c]) * mv;
  }
}

// ------------------------------ WMMA GEMM ----------------------------------
// Out[N,Cout] = A[N,K] @ W[K,Cout] (+bias, epilogue). 128 threads = 4 waves;
// block tile = 32 rows x 128 cols; each wave owns 32x32 (2x2 WMMA tiles),
// K stepped by 32. LDS holds bf16x2-packed tiles (B transposed so fragments
// are contiguous b128 loads).  N%32==0 and Cout%32==0 for every call site.
// EPI 0: bias + exact GELU + mask (h@w1). EPI 1: bias + mask + residual (h@w2).

template <int EPI>
__global__ __launch_bounds__(128) void gemm_wmma(
    const float* __restrict__ A, const float* __restrict__ W,
    const float* __restrict__ bias, const float* __restrict__ msk,
    const float* resid, float* Out, int N, int K, int Cout) {
  __shared__ __align__(16) unsigned Apk[32][20];   // [row][k-pair], padded
  __shared__ __align__(16) unsigned Bpk[128][20];  // [col][k-pair], padded
  const int tid  = threadIdx.x;
  const int wv   = tid >> 5;
  const int lane = tid & 31;
  const int r0   = blockIdx.x * 32;
  const int n0g  = blockIdx.y * 128;
  const int nw   = n0g + wv * 32;                  // wave column base
  const int mr   = lane & 15;
  const int kb2  = (lane >> 4) << 2;               // k-pair base: 0 / 4
  const bool active = (nw < Cout);                 // wave-uniform

  // cooperative staging of one K-step's tiles (all 128 threads participate)
  auto stage = [&](int k0) {
    // A tile: 32 rows x 16 bf16x2 pairs
    for (int p = tid; p < 32 * 16; p += 128) {
      int mm = p >> 4, kp = p & 15;
      const float* ap = &A[(size_t)(r0 + mm) * K + k0 + 2 * kp];
      Apk[mm][kp] = pack_bf2(ap[0], ap[1]);
    }
    // B tile transposed: 16 pairs x 128 cols -> Bpk[col][pair]
    for (int q = tid; q < 16 * 128; q += 128) {
      int kp = q >> 7, n = q & 127;
      int nc = n0g + n;
      float lo = 0.0f, hi = 0.0f;
      if (nc < Cout) {
        lo = W[(size_t)(k0 + 2 * kp) * Cout + nc];
        hi = W[(size_t)(k0 + 2 * kp + 1) * Cout + nc];
      }
      Bpk[n][kp] = pack_bf2(lo, hi);
    }
    // hint the next A tile into cache while we compute this one
    if (k0 + 32 < K && tid < 32)
      __builtin_prefetch(&A[(size_t)(r0 + tid) * K + k0 + 32], 0, 1);
  };

  if (!active) {
    // same barrier count as the active loop -> legal wave-uniform split
    for (int k0 = 0; k0 < K; k0 += 32) {
      __syncthreads();
      stage(k0);
      __syncthreads();
    }
    return;
  }

  v8f z = {0.f, 0.f, 0.f, 0.f, 0.f, 0.f, 0.f, 0.f};
  v8f acc00 = z, acc01 = z, acc10 = z, acc11 = z;

  for (int k0 = 0; k0 < K; k0 += 32) {
    __syncthreads();
    stage(k0);
    __syncthreads();
    v16bf a0 = frag(&Apk[mr][0],       kb2);            // rows r0 .. r0+15
    v16bf a1 = frag(&Apk[16 + mr][0],  kb2);            // rows r0+16 .. r0+31
    v16bf b0 = frag(&Bpk[wv * 32 + mr][0],      kb2);   // cols nw .. +15
    v16bf b1 = frag(&Bpk[wv * 32 + 16 + mr][0], kb2);   // cols nw+16 .. +31
    acc00 = __builtin_amdgcn_wmma_f32_16x16x32_bf16(
        false, a0, false, b0, (short)0, acc00, false, false);
    acc01 = __builtin_amdgcn_wmma_f32_16x16x32_bf16(
        false, a0, false, b1, (short)0, acc01, false, false);
    acc10 = __builtin_amdgcn_wmma_f32_16x16x32_bf16(
        false, a1, false, b0, (short)0, acc10, false, false);
    acc11 = __builtin_amdgcn_wmma_f32_16x16x32_bf16(
        false, a1, false, b1, (short)0, acc11, false, false);
  }

  // epilogue: bias + (GELU|residual) + mask, one 16x16 tile at a time
  #pragma unroll
  for (int ti = 0; ti < 2; ++ti) {
    #pragma unroll
    for (int tj = 0; tj < 2; ++tj) {
      const v8f acc = (ti == 0) ? (tj == 0 ? acc00 : acc01)
                                : (tj == 0 ? acc10 : acc11);
      const int ncol = nw + tj * 16 + (lane & 15);
      const float bn = bias[ncol];
      const int rb = r0 + ti * 16 + ((lane >> 4) << 3);  // lanes 16-31: rows m+8
      #pragma unroll
      for (int r = 0; r < 8; ++r) {
        int row = rb + r;
        float v = acc[r] + bn;
        float mv = msk[row];
        if (EPI == 0) {
          v = gelu_exact(v) * mv;
        } else {
          v = v * mv + resid[(size_t)row * Cout + ncol];
        }
        Out[(size_t)row * Cout + ncol] = v;
      }
    }
  }
}

// --------------------------------- GRN -------------------------------------

__global__ __launch_bounds__(256) void grn_zero(float* __restrict__ gx2, int C) {
  int c = blockIdx.x * 256 + threadIdx.x;
  if (c < C) gx2[c] = 0.0f;
}

__global__ __launch_bounds__(256) void grn_reduce(const float* __restrict__ H,
                                                  float* __restrict__ gx2,
                                                  int N, int C) {
  int c = blockIdx.y * 256 + threadIdx.x;
  if (c >= C) return;
  int r0 = blockIdx.x * 512;
  int r1 = r0 + 512; if (r1 > N) r1 = N;
  float acc = 0.0f;
  for (int r = r0; r < r1; ++r) {
    float v = H[(size_t)r * C + c];
    acc += v * v;
  }
  atomicAdd(&gx2[c], acc);
}

__global__ __launch_bounds__(256) void grn_finalize(const float* __restrict__ gx2,
                                                    float* __restrict__ nx, int C) {
  __shared__ float red[256];
  float s = 0.0f;
  for (int c = threadIdx.x; c < C; c += 256) s += sqrtf(gx2[c]);
  red[threadIdx.x] = s;
  __syncthreads();
  for (int o = 128; o > 0; o >>= 1) {
    if (threadIdx.x < o) red[threadIdx.x] += red[threadIdx.x + o];
    __syncthreads();
  }
  float mean = red[0] / (float)C;
  for (int c = threadIdx.x; c < C; c += 256)
    nx[c] = sqrtf(gx2[c]) / (mean + 1e-6f);
}

__global__ __launch_bounds__(256) void grn_apply(
    float* __restrict__ H, const float* __restrict__ nx,
    const float* __restrict__ g, const float* __restrict__ b,
    const float* __restrict__ m, int N, int C) {
  size_t idx = (size_t)blockIdx.x * 256 + threadIdx.x;
  size_t total = (size_t)N * C;
  if (idx >= total) return;
  int c = (int)(idx % C);
  int row = (int)(idx / C);
  float v = H[idx];
  H[idx] = (g[c] * (v * nx[c]) + b[c] + v) * m[row];
}

// --------------------------- downsample conv -------------------------------

__global__ __launch_bounds__(256) void ds_conv(
    const float* __restrict__ in, float* __restrict__ out,
    const float* __restrict__ w, const float* __restrict__ bias,
    const float* __restrict__ mnext, int Sin, int Cin, int Cout) {
  int Sout = Sin >> 1;
  size_t idx = (size_t)blockIdx.x * 256 + threadIdx.x;
  size_t total = (size_t)32 * Sout * Sout * Cout;
  if (idx >= total) return;
  int c = (int)(idx % Cout);
  int site = (int)(idx / Cout);
  int x = site % Sout, y = (site / Sout) % Sout, b = site / (Sout * Sout);
  float acc = bias[c];
  for (int ky = 0; ky < 2; ++ky)
    for (int kx = 0; kx < 2; ++kx) {
      const float* ip = &in[(((size_t)b * Sin + 2 * y + ky) * Sin + 2 * x + kx) * Cin];
      const float* wp = &w[(size_t)((ky * 2 + kx) * Cin) * Cout + c];
      for (int i = 0; i < Cin; ++i) acc += ip[i] * wp[(size_t)i * Cout];
    }
  out[idx] = acc * mnext[site];
}

// ------------------------- final NHWC -> NCHW ------------------------------

__global__ __launch_bounds__(256) void to_nchw_k(const float* __restrict__ in,
                                                 float* __restrict__ out) {
  int idx = blockIdx.x * 256 + threadIdx.x;
  if (idx >= 32 * 768 * 49) return;
  int x = idx % 7, y = (idx / 7) % 7, c = (idx / 49) % 768, b = idx / (49 * 768);
  out[idx] = in[(((size_t)b * 7 + y) * 7 + x) * 768 + c];
}

// ------------------------------- driver ------------------------------------

extern "C" void kernel_launch(void* const* d_in, const int* in_sizes, int n_in,
                              void* d_out, int out_size, void* d_ws, size_t ws_size,
                              hipStream_t stream) {
  (void)in_sizes; (void)n_in; (void)out_size; (void)ws_size;
  int idx = 0;
  const float* x        = (const float*)d_in[idx++];
  const int*   mask     = (const int*)d_in[idx++];
  const float* stem_w   = (const float*)d_in[idx++];
  const float* stem_b   = (const float*)d_in[idx++];
  const float* stem_lng = (const float*)d_in[idx++];
  const float* stem_lnb = (const float*)d_in[idx++];
  // downsample params: 3 x {ln_g, ln_b, w, b}
  const float* dsp[3][4];
  for (int i = 0; i < 3; ++i)
    for (int j = 0; j < 4; ++j) dsp[i][j] = (const float*)d_in[idx++];
  // stage blocks: {dw_k, dw_b, ln_g, ln_b, w1, b1, grn_g, grn_b, w2, b2}
  const int depths[4] = {3, 3, 9, 3};
  const float* blk[4][9][10];
  for (int i = 0; i < 4; ++i)
    for (int j = 0; j < depths[i]; ++j)
      for (int k = 0; k < 10; ++k) blk[i][j][k] = (const float*)d_in[idx++];

  // workspace layout (floats)
  float* ws   = (float*)d_ws;
  float* m56  = ws;                    // 32*56*56 = 100352
  float* m28  = m56 + 100352;          // 25088
  float* m14  = m28 + 25088;           // 6272
  float* m7   = m14 + 6272;            // 1568
  float* actA = m7 + 1568;             // up to 100352*96
  float* actB = actA + 9633792;
  float* actH = actB + 9633792;        // up to 100352*384
  float* gx2  = actH + 38535168;
  float* nxb  = gx2 + 4096;

  const float* msks[4] = {m56, m28, m14, m7};
  const int S[4] = {56, 28, 14, 7};
  const int D[4] = {96, 192, 384, 768};

  // masks
  build_mask_k<<<(100352 + 255) / 256, 256, 0, stream>>>(mask, m56);
  mask_down_k<<<(25088 + 255) / 256, 256, 0, stream>>>(m56, m28, 28);
  mask_down_k<<<(6272 + 255) / 256, 256, 0, stream>>>(m28, m14, 14);
  mask_down_k<<<(1568 + 255) / 256, 256, 0, stream>>>(m14, m7, 7);

  // stem: conv + LN + mask -> actA [32,56,56,96]
  stem_kernel<<<100352 / 8, 256, 0, stream>>>(x, stem_w, stem_b, stem_lng,
                                              stem_lnb, m56, actA);

  for (int i = 0; i < 4; ++i) {
    int N = 32 * S[i] * S[i];
    int d = D[i], d4 = 4 * d;
    if (i > 0) {
      int Np = 32 * S[i - 1] * S[i - 1];
      ln_mask<<<(Np + 7) / 8, 256, 0, stream>>>(actA, actB, dsp[i - 1][0],
                                                dsp[i - 1][1], msks[i - 1], Np,
                                                D[i - 1]);
      size_t tot = (size_t)N * d;
      ds_conv<<<(unsigned)((tot + 255) / 256), 256, 0, stream>>>(
          actB, actA, dsp[i - 1][2], dsp[i - 1][3], msks[i], S[i - 1], D[i - 1], d);
    }
    for (int j = 0; j < depths[i]; ++j) {
      const float* const* P = blk[i][j];
      size_t tot = (size_t)N * d;
      dwconv_mask<<<(unsigned)((tot + 255) / 256), 256, 0, stream>>>(
          actA, actB, P[0], P[1], msks[i], S[i], d);
      ln_mask<<<(N + 7) / 8, 256, 0, stream>>>(actB, actB, P[2], P[3], msks[i], N, d);

      dim3 g1(N / 32, (d4 + 127) / 128);
      gemm_wmma<0><<<g1, 128, 0, stream>>>(actB, P[4], P[5], msks[i], nullptr,
                                           actH, N, d, d4);

      grn_zero<<<(d4 + 255) / 256, 256, 0, stream>>>(gx2, d4);
      dim3 gr((N + 511) / 512, (d4 + 255) / 256);
      grn_reduce<<<gr, 256, 0, stream>>>(actH, gx2, N, d4);
      grn_finalize<<<1, 256, 0, stream>>>(gx2, nxb, d4);
      size_t totH = (size_t)N * d4;
      grn_apply<<<(unsigned)((totH + 255) / 256), 256, 0, stream>>>(
          actH, nxb, P[6], P[7], msks[i], N, d4);

      dim3 g2(N / 32, (d + 127) / 128);
      gemm_wmma<1><<<g2, 128, 0, stream>>>(actH, P[8], P[9], msks[i], actA,
                                           actA, N, d4, d);
    }
  }

  to_nchw_k<<<(32 * 768 * 49 + 255) / 256, 256, 0, stream>>>(actA, (float*)d_out);
}
